// Net_8486855377349
// MI455X (gfx1250) — matrix operand order repaired and checked by reading, not compiled, not run on hardware
//
#include <hip/hip_runtime.h>

typedef __attribute__((ext_vector_type(16))) _Float16 v16h;
typedef __attribute__((ext_vector_type(8)))  _Float16 v8h;
typedef __attribute__((ext_vector_type(8)))  float    v8f;
typedef __attribute__((ext_vector_type(4)))  float    v4f;

#define IN1 784
#define HDIM 64
#define KSTEPS 25                   // ceil(784/32), K padded to 800 with zeros
#define NTILES 4                    // 64 columns / 16
#define FRAGS (KSTEPS * NTILES)     // 100 B-fragments
#define FRAG_BYTES 1024             // 32 lanes * 16 halves * 2B
#define LDS_W1_BYTES (FRAGS * FRAG_BYTES)   // 102400
#define WAVES_PER_BLOCK 8
#define BLOCK_THREADS (WAVES_PER_BLOCK * 32)

// Straight-through threshold: forward value = soft + (hard - soft)
__device__ __forceinline__ float actv(float h, float n) {
    float t    = 10.0f * (h / n - 0.5f);
    float soft = 1.0f / (1.0f + __expf(-t));
    float hard = (h > 0.5f * n) ? 1.0f : 0.0f;
    return soft + (hard - soft);
}

extern "C" __global__ __launch_bounds__(BLOCK_THREADS)
void xornet_fused(const float* __restrict__ x, const float* __restrict__ w1,
                  const float* __restrict__ w2, float* __restrict__ out, int B)
{
    // 100KB of pre-swizzled f16 B-fragments holding (1 - 2*w1), plus sums/coeffs.
    // Identity: x + w - 2xw = x*(1-2w) + w  =>  h = x @ (1-2*w1)^T + sum_w1
    // so the WMMA alone produces h - sum_w1 (no sum_x needed at all).
    __shared__ __align__(16) unsigned char lds[LDS_W1_BYTES + 256 + 256 + 16];
    float* s_sumw1 = (float*)(lds + LDS_W1_BYTES);
    float* s_cw2   = (float*)(lds + LDS_W1_BYTES + 256);
    float* s_sumw2 = (float*)(lds + LDS_W1_BYTES + 512);

    const int tid = threadIdx.x;

    // ---- Stage (1 - 2*w1) into LDS as f16 WMMA B-fragments ----
    // B (32x16), contiguous K order used here: lane n (0-15) holds col n, K 0..15; lanes 16-31 hold K 16..31.
    for (int wi = tid; wi < FRAGS * 32; wi += BLOCK_THREADS) {
        int fi = wi >> 5;            // fragment index = kstep*4 + ntile
        int ln = wi & 31;            // lane slot inside fragment
        int kstep = fi >> 2, ntile = fi & 3;
        int col = ntile * 16 + (ln & 15);
        int ks  = kstep * 32 + ((ln & 16) ? 16 : 0);   // 16 contiguous K values
        v8h lo = {}, hi = {};
        if (ks < IN1) {              // runs of 16 are either fully valid or fully pad
            const float* p = w1 + col * IN1 + ks;
            v4f f0 = *(const v4f*)(p);
            v4f f1 = *(const v4f*)(p + 4);
            v4f f2 = *(const v4f*)(p + 8);
            v4f f3 = *(const v4f*)(p + 12);
            #pragma unroll
            for (int i = 0; i < 4; ++i) {
                lo[i]     = (_Float16)(1.0f - 2.0f * f0[i]);
                lo[i + 4] = (_Float16)(1.0f - 2.0f * f1[i]);
                hi[i]     = (_Float16)(1.0f - 2.0f * f2[i]);
                hi[i + 4] = (_Float16)(1.0f - 2.0f * f3[i]);
            }
        }
        *(v8h*)(lds + fi * FRAG_BYTES +       ln * 16) = lo;  // halves 0..7
        *(v8h*)(lds + fi * FRAG_BYTES + 512 + ln * 16) = hi;  // halves 8..15
    }

    if (tid < HDIM) {
        const float* p = w1 + tid * IN1;
        float s = 0.0f;
        for (int i = 0; i < IN1; ++i) s += p[i];
        s_sumw1[tid] = s;
        s_cw2[tid]   = 1.0f - 2.0f * w2[tid];   // fold layer-2 xor_linear the same way
    }
    if (tid == 0) {
        float s = 0.0f;
        for (int j = 0; j < HDIM; ++j) s += w2[j];
        *s_sumw2 = s;
    }
    __syncthreads();

    // ---- Per-wave 16-row tile ----
    const int lane    = tid & 31;
    const int wave    = tid >> 5;
    const int tile    = blockIdx.x * WAVES_PER_BLOCK + wave;
    const int rowBase = tile * 16;
    if (rowBase >= B) return;

    const int halfSel = (lane & 16) ? 1 : 0;
    const int r       = rowBase + (lane & 15);
    const float* xr   = x + (long)r * IN1;

    v8f cacc[NTILES] = {};

    for (int kstep = 0; kstep < KSTEPS; ++kstep) {
        // A (16x32), same contiguous K order as B: lanes 0-15 hold K 0-15; lanes 16-31 hold K 16-31
        int k0 = kstep * 32 + (halfSel ? 16 : 0);
        v4f a0 = {}, a1 = {};
        v4f a2 = {}, a3 = {};
        if (k0 < IN1) {              // only the K=784..799 pad region misses
            a0 = *(const v4f*)(xr + k0);
            a1 = *(const v4f*)(xr + k0 + 4);
            a2 = *(const v4f*)(xr + k0 + 8);
            a3 = *(const v4f*)(xr + k0 + 12);
        }

        v16h a;
        #pragma unroll
        for (int i = 0; i < 4; ++i) {
            a[i]      = (_Float16)a0[i];
            a[4 + i]  = (_Float16)a1[i];
            a[8 + i]  = (_Float16)a2[i];
            a[12 + i] = (_Float16)a3[i];
        }

        const unsigned char* fb = lds + (kstep * NTILES) * FRAG_BYTES;
        #pragma unroll
        for (int nt = 0; nt < NTILES; ++nt) {
            v8h b0 = *(const v8h*)(fb + nt * FRAG_BYTES +       lane * 16);
            v8h b1 = *(const v8h*)(fb + nt * FRAG_BYTES + 512 + lane * 16);
            v16h b;
            #pragma unroll
            for (int i = 0; i < 8; ++i) { b[i] = b0[i]; b[8 + i] = b1[i]; }
            cacc[nt] = __builtin_amdgcn_wmma_f32_16x16x32_f16(
                false, a, false, b, (short)0, cacc[nt], false, false);
        }
    }

    // cacc[nt][v] = (x @ (1-2*w1)^T)[row, col]; h = cacc + sum_w1[col]
    // Layer 1 activation fused with layer 2 dot product.
    float acc2[8] = {};
    #pragma unroll
    for (int nt = 0; nt < NTILES; ++nt) {
        int   colIdx = nt * 16 + (lane & 15);
        float sw1    = s_sumw1[colIdx];
        float cw     = s_cw2[colIdx];
        #pragma unroll
        for (int v = 0; v < 8; ++v) {
            float hv = cacc[nt][v] + sw1;
            float a1 = actv(hv, (float)IN1);
            acc2[v] += a1 * cw;
        }
    }
    // Reduce over the 16 lanes of each half-wave (each lane covered 4 columns)
    #pragma unroll
    for (int off = 1; off < 16; off <<= 1) {
        #pragma unroll
        for (int v = 0; v < 8; ++v)
            acc2[v] += __shfl_xor(acc2[v], off, 32);
    }

    const float sw2 = *s_sumw2;
    if ((lane & 15) == 0) {          // lane 0 -> rows 0..7, lane 16 -> rows 8..15
        v4f o0, o1;
        #pragma unroll
        for (int v = 0; v < 4; ++v) {
            o0[v] = actv(acc2[v]     + sw2, (float)HDIM);
            o1[v] = actv(acc2[v + 4] + sw2, (float)HDIM);
        }
        float* op = out + rowBase + (halfSel ? 8 : 0);
        *(v4f*)(op)     = o0;
        *(v4f*)(op + 4) = o1;
    }
}

extern "C" void kernel_launch(void* const* d_in, const int* in_sizes, int n_in,
                              void* d_out, int out_size, void* d_ws, size_t ws_size,
                              hipStream_t stream) {
    const float* x  = (const float*)d_in[0];
    const float* w1 = (const float*)d_in[1];
    const float* w2 = (const float*)d_in[2];
    float* out = (float*)d_out;
    int B      = in_sizes[0] / IN1;                 // 65536
    int tiles  = (B + 15) / 16;                     // 4096
    int blocks = (tiles + WAVES_PER_BLOCK - 1) / WAVES_PER_BLOCK;  // 512
    hipLaunchKernelGGL(xornet_fused, dim3(blocks), dim3(BLOCK_THREADS), 0, stream,
                       x, w1, w2, out, B);
}
